// FourierLearnerMoELayer_11828339933257
// MI455X (gfx1250) — matrix operand order repaired and checked by reading, not compiled, not run on hardware
//
#include <hip/hip_runtime.h>
#include <hip/hip_bf16.h>

typedef __attribute__((ext_vector_type(16))) __bf16 v16bf;
typedef __attribute__((ext_vector_type(8)))  float  v8f;

#define D_MODEL 512
#define T_SEQ   2048
#define B_BATCH 4
#define FF_DIM  2048
#define E_EXP   4
#define EPS_LN  1e-5f

static __device__ __forceinline__ unsigned short f32_bf16_rne(float f) {
    unsigned u = __float_as_uint(f);
    unsigned r = u + 0x7FFFu + ((u >> 16) & 1u);
    return (unsigned short)(r >> 16);
}

// Packed fragment layouts (per 16-wide tile, per 32-deep k-slab):
//   offset = ((tile16 * (K/32) + kq) * 32 + lane) * 16 + e     (16 ushorts/lane)
// A-operand element map (lane = hi*16+ml, ml = row%16):
//   e<8 : k = kq*32 + hi*8 + e        e>=8: k = kq*32 + 16 + hi*8 + (e-8)
// B-operand element map (lane = hi*16+nl, nl = col%16):
//   same as A: e<8 : k = kq*32 + hi*8 + e   e>=8: k = kq*32 + 16 + hi*8 + (e-8)
// Inverse map for A (k5 = k%32): hi = (k5>>3)&1 ; e = (k5&7) | ((k5>>4)<<3)

// ---------------------------------------------------------------- packers
__global__ void pack_a_f32(const float* __restrict__ src,
                           unsigned short* __restrict__ dst, int M, int Kd) {
    long long n = (long long)M * Kd;
    int kqc = Kd >> 5;
    long long i = (long long)blockIdx.x * blockDim.x + threadIdx.x;
    long long stride = (long long)gridDim.x * blockDim.x;
    for (; i < n; i += stride) {
        int e    = (int)(i & 15);
        int lane = (int)((i >> 4) & 31);
        long long rest = i >> 9;
        int kq   = (int)(rest % kqc);
        long long tm16 = rest / kqc;
        int hi = lane >> 4, ml = lane & 15;
        int k = kq * 32 + hi * 8 + ((e < 8) ? e : (16 + (e - 8)));
        long long row = tm16 * 16 + ml;
        dst[i] = f32_bf16_rne(src[row * Kd + k]);
    }
}

__global__ void pack_b_f32(const float* __restrict__ src,
                           unsigned short* __restrict__ dst, int Kd, int N) {
    long long n = (long long)Kd * N;
    int kqc = Kd >> 5;
    long long i = (long long)blockIdx.x * blockDim.x + threadIdx.x;
    long long stride = (long long)gridDim.x * blockDim.x;
    for (; i < n; i += stride) {
        int e    = (int)(i & 15);
        int lane = (int)((i >> 4) & 31);
        long long rest = i >> 9;
        int kq   = (int)(rest % kqc);
        long long tn = rest / kqc;
        int hi = lane >> 4, nl = lane & 15;
        int k = kq * 32 + hi * 8 + ((e < 8) ? e : (16 + (e - 8)));
        dst[i] = f32_bf16_rne(src[(long long)k * N + tn * 16 + nl]);
    }
}

// ---------------------------------------------------------------- WMMA GEMM
// A, B pre-packed bf16 fragments. Wave tile 64x32: 4 A-frags x 2 B-frags -> 8 WMMA.
// mode 0: C f32 = acc + bias
// mode 1: C = packed-A bf16 store of relu(acc+bias), packed for next-K = N
// mode 2: C f32 += rowscale[row*rsStride]*(acc+bias)
__global__ __launch_bounds__(256)
void gemm_bf16_wmma(const unsigned short* __restrict__ A, long long sA,
                    const unsigned short* __restrict__ B, long long sB,
                    void* __restrict__ Cv, int ldc, long long sC,
                    const float* __restrict__ bias,
                    const float* __restrict__ rowscale, int rsStride,
                    int M, int N, int Kd, int mode) {
    int bz = blockIdx.z;
    A += (long long)bz * sA;
    B += (long long)bz * sB;

    int tilesN = N >> 5;        // 32-wide n tiles
    int tilesM = M >> 6;        // 64-tall m tiles
    int tile = blockIdx.x * 8 + (threadIdx.x >> 5);
    if (tile >= tilesM * tilesN) return;
    int tm = tile / tilesN, tn = tile % tilesN;

    int lane = threadIdx.x & 31;
    int hi   = lane >> 4;
    int nl   = lane & 15;
    int kqc  = Kd >> 5;

    v8f acc[4][2] = {};

    const unsigned short* Bb0 = B + ((long long)(tn * 2)     * kqc * 32 + lane) * 16;
    const unsigned short* Bb1 = B + ((long long)(tn * 2 + 1) * kqc * 32 + lane) * 16;
    const unsigned short* Ab  = A + ((long long)(tm * 4)     * kqc * 32 + lane) * 16;

    for (int kq = 0; kq < kqc; ++kq) {
        if (kq + 1 < kqc) {   // hint next k-slab (global_prefetch_b8)
            __builtin_prefetch(Ab + ((long long)kq + 1) * 512, 0, 1);
            __builtin_prefetch(Bb0 + ((long long)kq + 1) * 512, 0, 1);
        }
        union { uint4 q[2]; v16bf v; } b0, b1;
        {
            const uint4* p0 = (const uint4*)(Bb0 + (long long)kq * 512);
            const uint4* p1 = (const uint4*)(Bb1 + (long long)kq * 512);
            b0.q[0] = p0[0]; b0.q[1] = p0[1];
            b1.q[0] = p1[0]; b1.q[1] = p1[1];
        }
#pragma unroll
        for (int s = 0; s < 4; ++s) {
            const uint4* ap = (const uint4*)(Ab + ((long long)s * kqc + kq) * 512);
            union { uint4 q[2]; v16bf v; } au;
            au.q[0] = ap[0];
            au.q[1] = ap[1];
            acc[s][0] = __builtin_amdgcn_wmma_f32_16x16x32_bf16(
                false, au.v, false, b0.v, (short)0, acc[s][0], false, false);
            acc[s][1] = __builtin_amdgcn_wmma_f32_16x16x32_bf16(
                false, au.v, false, b1.v, (short)0, acc[s][1], false, false);
        }
    }

    int m0 = tm << 6, n0 = tn << 5;
    float bv0 = bias ? bias[n0 + nl]      : 0.f;
    float bv1 = bias ? bias[n0 + 16 + nl] : 0.f;
    if (mode == 0) {
        float* C = (float*)Cv + (long long)bz * sC;
#pragma unroll
        for (int s = 0; s < 4; ++s)
#pragma unroll
            for (int j = 0; j < 8; ++j) {
                int row = m0 + s * 16 + hi * 8 + j;
                C[(long long)row * ldc + n0 + nl]      = acc[s][0][j] + bv0;
                C[(long long)row * ldc + n0 + 16 + nl] = acc[s][1][j] + bv1;
            }
    } else if (mode == 1) {
        // store relu(acc+bias) as bf16 in packed-A layout for a GEMM with K=N
        unsigned short* C = (unsigned short*)Cv;
        int kqcN = N >> 5;
        int k0 = n0 + nl, k1 = k0 + 16;
        int kq0 = k0 >> 5, k50 = k0 & 31;
        int kq1 = k1 >> 5, k51 = k1 & 31;
        int ph0 = (k50 >> 3) & 1, pe0 = (k50 & 7) | ((k50 >> 4) << 3);
        int ph1 = (k51 >> 3) & 1, pe1 = (k51 & 7) | ((k51 >> 4) << 3);
#pragma unroll
        for (int s = 0; s < 4; ++s) {
            long long tm16 = (long long)(m0 >> 4) + s;
#pragma unroll
            for (int j = 0; j < 8; ++j) {
                int ml = hi * 8 + j;
                C[(((tm16 * kqcN + kq0) * 32) + ph0 * 16 + ml) * 16 + pe0] =
                    f32_bf16_rne(fmaxf(acc[s][0][j] + bv0, 0.f));
                C[(((tm16 * kqcN + kq1) * 32) + ph1 * 16 + ml) * 16 + pe1] =
                    f32_bf16_rne(fmaxf(acc[s][1][j] + bv1, 0.f));
            }
        }
    } else {
        float* C = (float*)Cv + (long long)bz * sC;
#pragma unroll
        for (int s = 0; s < 4; ++s)
#pragma unroll
            for (int j = 0; j < 8; ++j) {
                int row = m0 + s * 16 + hi * 8 + j;
                float w = rowscale[(long long)row * rsStride];
                C[(long long)row * ldc + n0 + nl]      += w * (acc[s][0][j] + bv0);
                C[(long long)row * ldc + n0 + 16 + nl] += w * (acc[s][1][j] + bv1);
            }
    }
}

// ---------------------------------------------------------------- kv_sum
__global__ void kv_sum_kernel(const float* __restrict__ K,
                              const float* __restrict__ V,
                              float* __restrict__ kvs) {
    int idx = blockIdx.x * blockDim.x + threadIdx.x;
    if (idx >= B_BATCH * D_MODEL) return;
    int b = idx / D_MODEL, hd = idx % D_MODEL;
    const float* kp = K + (long long)b * T_SEQ * D_MODEL + hd;
    const float* vp = V + (long long)b * T_SEQ * D_MODEL + hd;
    float s = 0.f;
    for (int t = 0; t < T_SEQ; ++t)
        s += kp[(long long)t * D_MODEL] * vp[(long long)t * D_MODEL];
    kvs[idx] = s;
}

// wv = fbV + kvs broadcast, written as packed-A bf16 (K=512)
__global__ void add_kvsum_cvt(const float* __restrict__ fbV,
                              const float* __restrict__ kvs,
                              unsigned short* __restrict__ wvb) {
    long long n = (long long)B_BATCH * T_SEQ * D_MODEL;
    long long i = (long long)blockIdx.x * blockDim.x + threadIdx.x;
    long long stride = (long long)gridDim.x * blockDim.x;
    for (; i < n; i += stride) {
        int b  = (int)(i / ((long long)T_SEQ * D_MODEL));
        long long tok = i / D_MODEL;
        int hd = (int)(i % D_MODEL);
        float v = fbV[i] + kvs[b * D_MODEL + hd];
        long long tm16 = tok >> 4;
        int ml = (int)(tok & 15);
        int kq = hd >> 5, k5 = hd & 31;
        int phi = (k5 >> 3) & 1;
        int pe  = (k5 & 7) | ((k5 >> 4) << 3);
        wvb[(((tm16 * 16 + kq) * 32) + phi * 16 + ml) * 16 + pe] = f32_bf16_rne(v);
    }
}

// ---------------------------------------------------------------- LayerNorm
__global__ __launch_bounds__(256)
void residual_ln(const float* __restrict__ X, const float* __restrict__ Yadd,
                 const float* __restrict__ g, const float* __restrict__ bta,
                 float* __restrict__ Out, unsigned short* __restrict__ Outb) {
    int wave = threadIdx.x >> 5, lane = threadIdx.x & 31;
    long long tok = (long long)blockIdx.x * 8 + wave;
    const float* xp = X + tok * D_MODEL;
    const float* yp = Yadd + tok * D_MODEL;
    float v[16], s = 0.f, s2 = 0.f;
#pragma unroll
    for (int i = 0; i < 16; ++i) {
        int d = lane + 32 * i;
        float t = xp[d] + yp[d];
        v[i] = t; s += t; s2 += t * t;
    }
#pragma unroll
    for (int o = 16; o >= 1; o >>= 1) {
        s  += __shfl_xor(s, o, 32);
        s2 += __shfl_xor(s2, o, 32);
    }
    float mu  = s / (float)D_MODEL;
    float var = s2 / (float)D_MODEL - mu * mu;
    float inv = rsqrtf(var + EPS_LN);
    long long tm16 = tok >> 4;
    int ml = (int)(tok & 15);
#pragma unroll
    for (int i = 0; i < 16; ++i) {
        int d = lane + 32 * i;
        float o = (v[i] - mu) * inv * g[d] + bta[d];
        Out[tok * D_MODEL + d] = o;
        if (Outb) {
            int kq = d >> 5, k5 = d & 31;
            int phi = (k5 >> 3) & 1;
            int pe  = (k5 & 7) | ((k5 >> 4) << 3);
            Outb[(((tm16 * 16 + kq) * 32) + phi * 16 + ml) * 16 + pe] =
                f32_bf16_rne(o);
        }
    }
}

// ---------------------------------------------------------------- gating
__global__ __launch_bounds__(256)
void gate_kernel(const float* __restrict__ X, const float* __restrict__ Wg,
                 const float* __restrict__ bg, float* __restrict__ wout) {
    int wave = threadIdx.x >> 5, lane = threadIdx.x & 31;
    long long tok = (long long)blockIdx.x * 8 + wave;
    const float* xp = X + tok * D_MODEL;
    float acc[E_EXP] = {0.f, 0.f, 0.f, 0.f};
#pragma unroll
    for (int i = 0; i < 16; ++i) {
        int d = lane + 32 * i;
        float xv = xp[d];
#pragma unroll
        for (int e = 0; e < E_EXP; ++e) acc[e] += xv * Wg[d * E_EXP + e];
    }
#pragma unroll
    for (int e = 0; e < E_EXP; ++e)
#pragma unroll
        for (int o = 16; o >= 1; o >>= 1) acc[e] += __shfl_xor(acc[e], o, 32);
    if (lane == 0) {
        float lg[E_EXP], mx = -1e30f;
#pragma unroll
        for (int e = 0; e < E_EXP; ++e) { lg[e] = acc[e] + bg[e]; mx = fmaxf(mx, lg[e]); }
        float den = 0.f;
#pragma unroll
        for (int e = 0; e < E_EXP; ++e) { lg[e] = __expf(lg[e] - mx); den += lg[e]; }
#pragma unroll
        for (int e = 0; e < E_EXP; ++e) lg[e] /= den;
        int i0 = 0;
        for (int e = 1; e < E_EXP; ++e) if (lg[e] > lg[i0]) i0 = e;
        int i1 = -1;
        for (int e = 0; e < E_EXP; ++e)
            if (e != i0 && (i1 < 0 || lg[e] > lg[i1])) i1 = e;
        for (int e = 0; e < E_EXP; ++e)
            wout[tok * E_EXP + e] = (e == i0) ? lg[i0] : ((e == i1) ? lg[i1] : 0.f);
    }
}

// ================================================================ host side
extern "C" void kernel_launch(void* const* d_in, const int* in_sizes, int n_in,
                              void* d_out, int out_size, void* d_ws, size_t ws_size,
                              hipStream_t stream) {
    (void)in_sizes; (void)n_in; (void)out_size; (void)ws_size;
    const float* x    = (const float*)d_in[0];
    const float* fb   = (const float*)d_in[1];
    const float* Wk   = (const float*)d_in[2];
    const float* bk   = (const float*)d_in[3];
    const float* Wv   = (const float*)d_in[4];
    const float* bv   = (const float*)d_in[5];
    const float* Wo   = (const float*)d_in[6];
    const float* bo   = (const float*)d_in[7];
    const float* ln1g = (const float*)d_in[8];
    const float* ln1b = (const float*)d_in[9];
    const float* Wg   = (const float*)d_in[10];
    const float* bg   = (const float*)d_in[11];
    const float* W1   = (const float*)d_in[12];
    const float* b1   = (const float*)d_in[13];
    const float* W2   = (const float*)d_in[14];
    const float* b2   = (const float*)d_in[15];
    const float* ln2g = (const float*)d_in[16];
    const float* ln2b = (const float*)d_in[17];
    float* out = (float*)d_out;

    const long long NTOK = (long long)B_BATCH * T_SEQ;           // 8192
    const long long ND   = NTOK * D_MODEL;                       // 4M
    const long long NFB  = (long long)B_BATCH * T_SEQ * T_SEQ;   // 16.7M
    const long long NW12 = (long long)E_EXP * D_MODEL * FF_DIM;  // 4M

    char* ws = (char*)d_ws;
    size_t off = 0;
    auto alloc = [&](size_t bytes) -> void* {
        void* p = ws + off;
        off = (off + bytes + 255) & ~(size_t)255;
        return p;
    };
    unsigned short* xb   = (unsigned short*)alloc(ND * 2);                       // packed A
    unsigned short* Wkb  = (unsigned short*)alloc((size_t)D_MODEL * D_MODEL * 2);// packed B
    unsigned short* Wvb  = (unsigned short*)alloc((size_t)D_MODEL * D_MODEL * 2);
    unsigned short* Wob  = (unsigned short*)alloc((size_t)D_MODEL * D_MODEL * 2);
    unsigned short* W1b  = (unsigned short*)alloc(NW12 * 2);
    unsigned short* W2b  = (unsigned short*)alloc(NW12 * 2);
    unsigned short* fbb  = (unsigned short*)alloc(NFB * 2);                      // packed A
    float*          Kf   = (float*)alloc(ND * 4);
    float*          Vf   = (float*)alloc(ND * 4);
    unsigned short* Vb   = (unsigned short*)alloc(ND * 2);                       // packed B
    float*          kvs  = (float*)alloc((size_t)B_BATCH * D_MODEL * 4);
    float*          fbV  = (float*)alloc(ND * 4);
    unsigned short* wvb  = (unsigned short*)alloc(ND * 2);                       // packed A
    float*          proj = (float*)alloc(ND * 4);
    float*          x1   = (float*)alloc(ND * 4);
    unsigned short* x1b  = (unsigned short*)alloc(ND * 2);                       // packed A
    float*          wgt  = (float*)alloc((size_t)NTOK * E_EXP * 4);
    unsigned short* he   = (unsigned short*)alloc((size_t)NTOK * FF_DIM * 2);    // packed A
    float*          ff   = (float*)alloc(ND * 4);

    dim3 blk(256);
    // -------- pack operands (f32 -> bf16 fragment layout)
    pack_a_f32<<<2048, blk, 0, stream>>>(x,  xb,  8192, 512);
    pack_a_f32<<<4096, blk, 0, stream>>>(fb, fbb, 8192, 2048);   // 4x[2048,2048] contiguous
    pack_b_f32<<<256,  blk, 0, stream>>>(Wk, Wkb, 512, 512);
    pack_b_f32<<<256,  blk, 0, stream>>>(Wv, Wvb, 512, 512);
    pack_b_f32<<<256,  blk, 0, stream>>>(Wo, Wob, 512, 512);
    for (int e = 0; e < E_EXP; ++e) {
        pack_b_f32<<<1024, blk, 0, stream>>>(W1 + (long long)e * D_MODEL * FF_DIM,
                                             W1b + (long long)e * D_MODEL * FF_DIM,
                                             512, 2048);
        pack_b_f32<<<1024, blk, 0, stream>>>(W2 + (long long)e * FF_DIM * D_MODEL,
                                             W2b + (long long)e * FF_DIM * D_MODEL,
                                             2048, 512);
    }

    // -------- K = x@Wk + bk ; V = x@Wv + bv   [8192,512] f32
    {
        int tiles = (8192 / 64) * (512 / 32);
        dim3 g(tiles / 8, 1, 1);
        gemm_bf16_wmma<<<g, blk, 0, stream>>>(xb, 0, Wkb, 0, Kf, D_MODEL, 0,
            bk, nullptr, 0, 8192, 512, 512, 0);
        gemm_bf16_wmma<<<g, blk, 0, stream>>>(xb, 0, Wvb, 0, Vf, D_MODEL, 0,
            bv, nullptr, 0, 8192, 512, 512, 0);
    }
    kv_sum_kernel<<<(B_BATCH * D_MODEL + 255) / 256, blk, 0, stream>>>(Kf, Vf, kvs);
    for (int b = 0; b < B_BATCH; ++b)
        pack_b_f32<<<1024, blk, 0, stream>>>(Vf + (long long)b * T_SEQ * D_MODEL,
                                             Vb + (long long)b * T_SEQ * D_MODEL,
                                             2048, 512);

    // -------- fbV[b] = fb[b] @ V[b]   batched [2048,2048]@[2048,512]
    {
        int tiles = (2048 / 64) * (512 / 32);
        dim3 g(tiles / 8, 1, B_BATCH);
        gemm_bf16_wmma<<<g, blk, 0, stream>>>(
            fbb, (long long)T_SEQ * T_SEQ,
            Vb,  (long long)T_SEQ * D_MODEL,
            fbV, D_MODEL, (long long)T_SEQ * D_MODEL,
            nullptr, nullptr, 0, 2048, 512, 2048, 0);
    }
    add_kvsum_cvt<<<2048, blk, 0, stream>>>(fbV, kvs, wvb);

    // -------- proj = wv @ Wo + bo ; x1 = LN(x + proj)
    {
        int tiles = (8192 / 64) * (512 / 32);
        dim3 g(tiles / 8, 1, 1);
        gemm_bf16_wmma<<<g, blk, 0, stream>>>(wvb, 0, Wob, 0, proj, D_MODEL, 0,
            bo, nullptr, 0, 8192, 512, 512, 0);
    }
    residual_ln<<<NTOK / 8, blk, 0, stream>>>(x, proj, ln1g, ln1b, x1, x1b);

    // -------- gating
    gate_kernel<<<NTOK / 8, blk, 0, stream>>>(x1, Wg, bg, wgt);

    // -------- MoE FFN (all experts, matching reference; w is 0 for non-top2)
    hipMemsetAsync(ff, 0, ND * 4, stream);
    for (int e = 0; e < E_EXP; ++e) {
        const unsigned short* W1e = W1b + (long long)e * D_MODEL * FF_DIM;
        const unsigned short* W2e = W2b + (long long)e * FF_DIM * D_MODEL;
        const float* b1e = b1 + (long long)e * FF_DIM;
        const float* b2e = b2 + (long long)e * D_MODEL;
        {   // he = relu(x1 @ W1e + b1e), stored packed-A for K=2048
            int tiles = (8192 / 64) * (2048 / 32);
            dim3 g(tiles / 8, 1, 1);
            gemm_bf16_wmma<<<g, blk, 0, stream>>>(x1b, 0, W1e, 0, he, FF_DIM, 0,
                b1e, nullptr, 0, 8192, 2048, 512, 1);
        }
        {   // ff += w[:,e] * (he @ W2e + b2e)
            int tiles = (8192 / 64) * (512 / 32);
            dim3 g(tiles / 8, 1, 1);
            gemm_bf16_wmma<<<g, blk, 0, stream>>>(he, 0, W2e, 0, ff, D_MODEL, 0,
                b2e, wgt + e, E_EXP, 8192, 512, 2048, 2);
        }
    }

    // -------- out = LN(x1 + ff)
    residual_ln<<<NTOK / 8, blk, 0, stream>>>(x1, ff, ln2g, ln2b, out, nullptr);
}
